// ModernNCA_60730837566126
// MI455X (gfx1250) — compile-verified, matrix operand order checked by reading, not run on hardware
//
#include <hip/hip_runtime.h>
#include <stdint.h>

typedef __attribute__((ext_vector_type(16))) __bf16 v16bf;
typedef __attribute__((ext_vector_type(8)))  __bf16 v8bf;
typedef __attribute__((ext_vector_type(8)))  float  v8f;
typedef unsigned int tdm_u32x4 __attribute__((ext_vector_type(4)));
typedef int          tdm_i32x8 __attribute__((ext_vector_type(8)));
typedef int          tdm_i32x4 __attribute__((ext_vector_type(4)));

union Frag { v16bf v; v8bf h[2]; };

constexpr int kB = 1024, kN = 16384, kNNUM = 64, kNFREQ = 48, kDEMB = 16, kC = 10;
constexpr int kH1 = 1024, kH2 = 512, kDOUT = kNNUM * kDEMB;   // 1024
constexpr int kRows = kB + kN;                                 // 17408
constexpr float kTwoPi = 6.28318530717958647692f;

// ---- workspace layout (bytes). S overlays e+h1 after they are consumed. ----
constexpr size_t OFF_W1T = 0;                                      // 1024x1024 bf16 = 2 MB
constexpr size_t OFF_W2T = OFF_W1T + (size_t)kH1 * kDOUT * 2;      // 512x1024 bf16 = 1 MB
constexpr size_t OFF_KN  = OFF_W2T + (size_t)kH2 * kH1 * 2;        // 16384 f32
constexpr size_t OFF_H2  = OFF_KN  + (size_t)kN * 4;               // 17408x512 bf16
constexpr size_t OFF_E   = OFF_H2  + (size_t)kRows * kH2 * 2;      // 17408x1024 bf16
constexpr size_t OFF_H1  = OFF_E   + (size_t)kRows * kDOUT * 2;    // 17408x1024 bf16
constexpr size_t OFF_S   = OFF_E;                                  // 1024x16384 f32 (overlay)

// LDS tile geometry: padded row stride 80 B (32 bf16 = 64 B + 16 B pad)
constexpr int LDS_ROW_B   = 80;
constexpr int LDS_A_BUF   = 64  * LDS_ROW_B;   // 5120 B
constexpr int LDS_B_BUF   = 128 * LDS_ROW_B;   // 10240 B
constexpr int LDS_A_OFF   = 0;                 // 2 buffers: 0, 5120
constexpr int LDS_B_OFF   = 2 * LDS_A_BUF;     // 2 buffers: 10240, 20480
constexpr int LDS_TOTAL   = LDS_B_OFF + 2 * LDS_B_BUF;  // 30720 B

// ---------------- TDM issue helper ----------------
// D# per CDNA5 ISA ch8: group0 = {flags(count=1), lds_addr, ga_lo, ga_hi|type=2}
// group1: data_size=2B, pad_enable, pad_interval=16DW, pad_amount=4DW,
//         tensor_dim0=32, tensor_dim1=rows, tile_dim0=32, tile_dim1=rows,
//         tensor_dim0_stride=K (elements). 2D tile -> groups 2/3 zero (NULL).
__device__ __forceinline__ void tdm_tile_load(const __bf16* gptr, uint32_t lds_off,
                                              int rows, int K) {
    uint64_t ga = (uint64_t)(uintptr_t)gptr;
    tdm_u32x4 g0 = { 1u,                                   // count=1, user desc
                     lds_off,                              // lds_addr (bytes)
                     (uint32_t)ga,                         // global_addr[31:0]
                     (uint32_t)(ga >> 32) | (2u << 30) };  // ga[56:32] | type=2
    tdm_i32x8 g1 = { (int)((1u << 16) | (1u << 20) | (3u << 22) | (3u << 25)),
                     (int)(32u << 16),                     // tensor_dim0[15:0]=32
                     (int)((uint32_t)rows << 16),          // tensor_dim1[15:0]=rows
                     (int)(32u << 16),                     // tile_dim0=32
                     rows,                                 // tile_dim1=rows
                     K,                                    // dim0_stride[31:0]
                     0, 0 };
    tdm_i32x4 z4 = {0, 0, 0, 0};
#if defined(__clang_major__) && __clang_major__ >= 23
    tdm_i32x8 z8 = {0, 0, 0, 0, 0, 0, 0, 0};
    __builtin_amdgcn_tensor_load_to_lds(g0, g1, z4, z4, z8, 0);
#else
    __builtin_amdgcn_tensor_load_to_lds(g0, g1, z4, z4, 0);
#endif
}

// ---------------- weight transpose + f32 -> bf16 ----------------
__global__ void k_transpose_bf16(const float* __restrict__ W, __bf16* __restrict__ Wt,
                                 int K, int N) {
    long idx = (long)blockIdx.x * blockDim.x + threadIdx.x;
    if (idx >= (long)K * N) return;
    int k = (int)(idx / N), n = (int)(idx % N);
    Wt[(size_t)n * K + k] = (__bf16)W[idx];
}

// ---------------- PLR periodic encoder: one block per row ----------------
__global__ __launch_bounds__(256)
void k_encode(const float* __restrict__ x_num, const float* __restrict__ cand_num,
              const float* __restrict__ freqs, const float* __restrict__ W_num,
              const float* __restrict__ b_num, __bf16* __restrict__ e) {
    __shared__ float sx[kNNUM];
    __shared__ float sp[kNNUM * 2 * kNFREQ];   // 64 x 96 f32 = 24 KB
    int r = blockIdx.x;
    int t = threadIdx.x;
    const float* xr = (r < kB) ? (x_num + (size_t)r * kNNUM)
                               : (cand_num + (size_t)(r - kB) * kNNUM);
    if (t < kNNUM) sx[t] = xr[t];
    __syncthreads();
    for (int i = t; i < kNNUM * 2 * kNFREQ; i += 256) {
        int n  = i / (2 * kNFREQ);
        int f2 = i % (2 * kNFREQ);
        int f  = (f2 < kNFREQ) ? f2 : (f2 - kNFREQ);
        float v = kTwoPi * sx[n] * freqs[n * kNFREQ + f];
        sp[i] = (f2 < kNFREQ) ? __cosf(v) : __sinf(v);
    }
    __syncthreads();
    for (int o = t; o < kNNUM * kDEMB; o += 256) {
        int n = o / kDEMB, d = o % kDEMB;
        float sum = b_num[o];
        const float* wn = W_num + ((size_t)n * (2 * kNFREQ)) * kDEMB + d;
        const float* pp = sp + n * (2 * kNFREQ);
#pragma unroll 8
        for (int f = 0; f < 2 * kNFREQ; ++f)
            sum = fmaf(pp[f], wn[(size_t)f * kDEMB], sum);
        e[(size_t)r * kDOUT + o] = (__bf16)fmaxf(sum, 0.0f);
    }
}

// ---------------- bf16 WMMA GEMM: C = A[M,K] * Bt[Nt,K]^T (+bias, relu) ----------------
// Block: 256 thr = 8 waves; wave tile 32x32 (2x2 wmma); block tile 64 x 128.
// TDM double-buffers the 64x32 A tile and 128x32 B tile into padded LDS.
// NOTE: the TDM builtin takes the LDS address as integer data, so the compiler
// cannot see the LDS writes. A dynamically-dead guard store (K < 0 is never
// true, but K is a runtime argument) keeps the LDS object "may-be-written" so
// the fragment reads stay ordinary ds_load_b128 instead of folding to undef.
__global__ __launch_bounds__(256)
void k_gemm_bt(const __bf16* __restrict__ A, const __bf16* __restrict__ Bt,
               const float* __restrict__ bias, __bf16* __restrict__ Cb,
               float* __restrict__ Cf, int M, int Nt, int K, int relu_bf16) {
    __shared__ __align__(16) char smem[LDS_TOTAL];

    if (K < 0) {   // never taken; defeats "smem never stored" analysis
        for (int i = threadIdx.x; i < LDS_TOTAL; i += 256) smem[i] = (char)K;
    }

    int lane = threadIdx.x & 31;
    int w    = threadIdx.x >> 5;
    int wm = w & 1, wn = w >> 1;
    int lr = lane & 15;        // row/col within 16-tile
    int g  = lane >> 4;        // half-lane group -> K phase
    int row0 = blockIdx.y * 64  + wm * 32;
    int col0 = blockIdx.x * 128 + wn * 32;

    const __bf16* Atile = A  + (size_t)(blockIdx.y * 64)  * K;
    const __bf16* Btile = Bt + (size_t)(blockIdx.x * 128) * K;
    int nk = K >> 5;

    // Prologue: prime both pipeline stages from wave 0 only (TDM ignores EXEC,
    // descriptors are uniform SGPR data; TENSORcnt is per-wave).
    if (threadIdx.x < 32) {
        tdm_tile_load(Atile, LDS_A_OFF, 64, K);
        tdm_tile_load(Btile, LDS_B_OFF, 128, K);
        if (nk > 1) {
            tdm_tile_load(Atile + 32, LDS_A_OFF + LDS_A_BUF, 64, K);
            tdm_tile_load(Btile + 32, LDS_B_OFF + LDS_B_BUF, 128, K);
        }
    }

    v8f acc[2][2] = {};
    for (int s = 0; s < nk; ++s) {
        int cur = s & 1;
        // SOPP wait executes per-wave regardless of EXEC; waves 1..7 have
        // TENSORcnt==0 so this is free for them.
        if (s + 1 < nk) __builtin_amdgcn_s_wait_tensorcnt(2);  // stage `cur` done
        else            __builtin_amdgcn_s_wait_tensorcnt(0);  // drain last stage
        asm volatile("" ::: "memory");   // LDS changed behind compiler's back
        __syncthreads();                 // publish LDS buffer `cur` to all waves

        const char* abase = smem + LDS_A_OFF + cur * LDS_A_BUF;
        const char* bbase = smem + LDS_B_OFF + cur * LDS_B_BUF;
        Frag a[2], b[2];
#pragma unroll
        for (int i = 0; i < 2; ++i) {
            const char* p = abase + (wm * 32 + i * 16 + lr) * LDS_ROW_B + g * 16;
            a[i].h[0] = *(const v8bf*)p;          // K = g*8 + 0..7
            a[i].h[1] = *(const v8bf*)(p + 32);   // K = 16 + g*8 + 0..7
        }
#pragma unroll
        for (int j = 0; j < 2; ++j) {
            const char* p = bbase + (wn * 32 + j * 16 + lr) * LDS_ROW_B + g * 16;
            b[j].h[0] = *(const v8bf*)p;
            b[j].h[1] = *(const v8bf*)(p + 32);
        }
#pragma unroll
        for (int i = 0; i < 2; ++i)
#pragma unroll
            for (int j = 0; j < 2; ++j)
                acc[i][j] = __builtin_amdgcn_wmma_f32_16x16x32_bf16(
                    false, a[i].v, false, b[j].v, (short)0, acc[i][j], false, false);

        __syncthreads();   // all waves done reading buffer `cur`
        if (threadIdx.x < 32 && s + 2 < nk) {
            int k2 = (s + 2) * 32;
            tdm_tile_load(Atile + k2, LDS_A_OFF + cur * LDS_A_BUF, 64, K);
            tdm_tile_load(Btile + k2, LDS_B_OFF + cur * LDS_B_BUF, 128, K);
        }
    }

#pragma unroll
    for (int i = 0; i < 2; ++i)
#pragma unroll
        for (int j = 0; j < 2; ++j) {
            int n = col0 + j * 16 + lr;
            float bs = bias ? bias[n] : 0.0f;
#pragma unroll
            for (int tt = 0; tt < 8; ++tt) {
                int m = row0 + i * 16 + g * 8 + tt;     // C layout: vgpr tt -> M row
                float vv = acc[i][j][tt] + bs;
                if (relu_bf16) Cb[(size_t)m * Nt + n] = (__bf16)fmaxf(vv, 0.0f);
                else           Cf[(size_t)m * Nt + n] = vv;
            }
        }
}

// ---------------- ||k_j||^2 : one wave per candidate row ----------------
__global__ __launch_bounds__(256)
void k_rownorm(const __bf16* __restrict__ kmat, float* __restrict__ kn) {
    int gw   = (blockIdx.x * 256 + threadIdx.x) >> 5;
    int lane = threadIdx.x & 31;
    if (gw >= kN) return;
    const __bf16* row = kmat + (size_t)gw * kH2;
    float s = 0.0f;
    for (int d = lane; d < kH2; d += 32) { float v = (float)row[d]; s = fmaf(v, v, s); }
    for (int off = 16; off > 0; off >>= 1) s += __shfl_xor(s, off, 32);
    if (lane == 0) kn[gw] = s;
}

// ---------------- softmax(2s - ||k||^2) + class aggregation: one block per query ----------------
__global__ __launch_bounds__(256)
void k_nca(const float* __restrict__ S, const float* __restrict__ kn,
           const int* __restrict__ y, float* __restrict__ out) {
    __shared__ float red[256];
    int i = blockIdx.x, t = threadIdx.x;
    const float* Si = S + (size_t)i * kN;

    float lmax = -3.4e38f;
    for (int j = t; j < kN; j += 256)
        lmax = fmaxf(lmax, 2.0f * Si[j] - kn[j]);
    red[t] = lmax; __syncthreads();
    for (int off = 128; off > 0; off >>= 1) {
        if (t < off) red[t] = fmaxf(red[t], red[t + off]);
        __syncthreads();
    }
    float m = red[0]; __syncthreads();

    float lsum = 0.0f, cls[kC];
#pragma unroll
    for (int c = 0; c < kC; ++c) cls[c] = 0.0f;
    for (int j = t; j < kN; j += 256) {
        float l = 2.0f * Si[j] - kn[j];
        float ev = __expf(l - m);
        lsum += ev;
        int yc = y[j];
#pragma unroll
        for (int c = 0; c < kC; ++c) cls[c] += (yc == c) ? ev : 0.0f;
    }
    red[t] = lsum; __syncthreads();
    for (int off = 128; off > 0; off >>= 1) { if (t < off) red[t] += red[t + off]; __syncthreads(); }
    float inv = 1.0f / red[0]; __syncthreads();

    for (int c = 0; c < kC; ++c) {
        red[t] = cls[c]; __syncthreads();
        for (int off = 128; off > 0; off >>= 1) { if (t < off) red[t] += red[t + off]; __syncthreads(); }
        if (t == 0) out[(size_t)i * kC + c] = red[0] * inv;
        __syncthreads();
    }
}

extern "C" void kernel_launch(void* const* d_in, const int* in_sizes, int n_in,
                              void* d_out, int out_size, void* d_ws, size_t ws_size,
                              hipStream_t stream) {
    (void)in_sizes; (void)n_in; (void)out_size; (void)ws_size;
    const float* x_num  = (const float*)d_in[0];
    const float* cand   = (const float*)d_in[1];
    const int*   cand_y = (const int*)d_in[2];
    const float* freqs  = (const float*)d_in[3];
    const float* W_num  = (const float*)d_in[4];
    const float* b_num  = (const float*)d_in[5];
    const float* W1     = (const float*)d_in[6];
    const float* b1     = (const float*)d_in[7];
    const float* W2     = (const float*)d_in[8];
    const float* b2     = (const float*)d_in[9];
    float* out = (float*)d_out;

    char* ws = (char*)d_ws;
    __bf16* w1t = (__bf16*)(ws + OFF_W1T);
    __bf16* w2t = (__bf16*)(ws + OFF_W2T);
    float*  kn  = (float*) (ws + OFF_KN);
    __bf16* h2  = (__bf16*)(ws + OFF_H2);
    __bf16* e   = (__bf16*)(ws + OFF_E);
    __bf16* h1  = (__bf16*)(ws + OFF_H1);
    float*  S   = (float*) (ws + OFF_S);

    // W1 [DOUT,H1] -> w1t [H1,DOUT]; W2 [H1,H2] -> w2t [H2,H1] (bf16)
    k_transpose_bf16<<<(kDOUT * kH1) / 256, 256, 0, stream>>>(W1, w1t, kDOUT, kH1);
    k_transpose_bf16<<<(kH1 * kH2) / 256, 256, 0, stream>>>(W2, w2t, kH1, kH2);

    // encoder for all 17408 rows (queries then candidates)
    k_encode<<<kRows, 256, 0, stream>>>(x_num, cand, freqs, W_num, b_num, e);

    // h1 = relu(e @ W1 + b1)        [17408,1024]
    k_gemm_bt<<<dim3(kH1 / 128, kRows / 64), 256, 0, stream>>>(
        e, w1t, b1, h1, nullptr, kRows, kH1, kDOUT, 1);
    // h2 = relu(h1 @ W2 + b2)       [17408,512]
    k_gemm_bt<<<dim3(kH2 / 128, kRows / 64), 256, 0, stream>>>(
        h1, w2t, b2, h2, nullptr, kRows, kH2, kH1, 1);

    // ||k_j||^2
    k_rownorm<<<(kN * 32) / 256, 256, 0, stream>>>(h2 + (size_t)kB * kH2, kn);

    // S = q @ k^T  (k already stored row-major [N,512] == Bt form)   [1024,16384] f32
    k_gemm_bt<<<dim3(kN / 128, kB / 64), 256, 0, stream>>>(
        h2, h2 + (size_t)kB * kH2, nullptr, nullptr, S, kB, kN, kH2, 0);

    // softmax over 2S - ||k||^2, aggregate one-hot labels -> probs [1024,10]
    k_nca<<<kB, 256, 0, stream>>>(S, kn, cand_y, out);
}
